// GeneralizedRingAttractorNoGain_37297495998725
// MI455X (gfx1250) — matrix operand order repaired and checked by reading, compiled, not run on hardware
//
#include <hip/hip_runtime.h>

// ---------------- problem constants ----------------
#define NN    256      // neurons
#define TT    128      // timesteps
#define AD    32       // action dim
#define NA    33       // AD + 1 (Wo folded in as action 32 with coeff J1)
#define BB    64       // batch
#define J0C   (-0.1f)
#define J1C   (0.1f)
#define ALPHAC 0.15f

// ---------------- mapping constants ----------------
#define WGS   512      // 16 wave32 per workgroup
#define NWG   32       // 16 m-slices x 2 batch-halves
#define MCOLS 16       // output columns per workgroup
#define BLOC  32       // batches per workgroup
#define RSTR  264      // padded LDS row stride (bf16 elems): 256 + 8 -> conflict-free b128 loads

typedef __attribute__((ext_vector_type(16))) __bf16   v16bf;
typedef __attribute__((ext_vector_type(8)))  float    v8f;
typedef __attribute__((ext_vector_type(4)))  unsigned v4u;

union ABFrag { v4u u[2]; v16bf v; };

// LDS layout (bytes)
constexpr size_t OFF_W   = 0;                         // weights: [NA*MCOLS][RSTR] bf16
constexpr size_t SZ_W    = (size_t)NA * MCOLS * RSTR * 2;   // 278784
constexpr size_t OFF_R   = OFF_W + SZ_W;              // r tile: [BLOC][RSTR] bf16
constexpr size_t SZ_R    = (size_t)BLOC * RSTR * 2;         // 16896
constexpr size_t OFF_P   = OFF_R + SZ_R;              // partials: [2 tiles][8 ag][16][16] f32
constexpr size_t SZ_P    = 4096 * 4;                        // 16384
constexpr size_t OFF_C   = OFF_P + SZ_P;              // coeffs: [BLOC][NA] f32
constexpr size_t SZ_C    = (size_t)BLOC * NA * 4;           // 4224
constexpr size_t OFF_S   = OFF_C + SZ_C;              // row sums: [BLOC] f32
constexpr size_t SZ_S    = 128;
constexpr size_t OFF_RED = OFF_S + SZ_S;              // sum-reduce scratch: [BLOC*16] f32
constexpr size_t SZ_RED  = 512 * 4;
constexpr size_t SMEM_BYTES = OFF_RED + SZ_RED;       // 318464  (<= 320KB WGP LDS)

__device__ __forceinline__ unsigned short f2bf(float f) {
  unsigned u = __builtin_bit_cast(unsigned, f);
  u += 0x7fffu + ((u >> 16) & 1u);                    // round-to-nearest-even
  return (unsigned short)(u >> 16);
}
__device__ __forceinline__ float bf2f(unsigned short h) {
  return __builtin_bit_cast(float, (unsigned)h << 16);
}

__device__ __forceinline__ void grid_sync(unsigned* cnt, unsigned expected) {
  __threadfence();                                    // make r-writes device-visible
  __syncthreads();
  if (threadIdx.x == 0) {
    __hip_atomic_fetch_add(cnt, 1u, __ATOMIC_ACQ_REL, __HIP_MEMORY_SCOPE_AGENT);
    while (__hip_atomic_load(cnt, __ATOMIC_ACQUIRE, __HIP_MEMORY_SCOPE_AGENT) < expected)
      __builtin_amdgcn_s_sleep(2);
  }
  __syncthreads();
}

extern "C" __global__ __launch_bounds__(WGS, 1)
void ring_attractor_kernel(const float* __restrict__ action,
                           const float* __restrict__ Wo,
                           const float* __restrict__ Wa,
                           float* __restrict__ out,
                           unsigned* __restrict__ counter) {
  extern __shared__ char smem[];
  unsigned short* W_lds   = (unsigned short*)(smem + OFF_W);
  unsigned short* r_lds   = (unsigned short*)(smem + OFF_R);
  float*          p_lds   = (float*)(smem + OFF_P);
  float*          c_lds   = (float*)(smem + OFF_C);
  float*          s_lds   = (float*)(smem + OFF_S);
  float*          red_lds = (float*)(smem + OFF_RED);

  float* hist  = out;                                  // [B][T][N] r_history
  float* bump  = out + (size_t)BB * TT * NN;           // [B][T][N] bump (doubles as r state)
  float* r0scr = hist;                                 // first 64*256 floats: r0 scratch (overwritten later)

  const int g   = blockIdx.x;
  const int m0  = (g & 15) * MCOLS;                    // column slice
  const int b0  = (g >> 4) * BLOC;                     // batch half
  const int tid = threadIdx.x;

  const int wave  = tid >> 5, lane = tid & 31;
  const int tile  = wave & 1;                          // 16-batch sub-tile within the half
  const int ag    = wave >> 1;                         // a-group 0..7 (7 takes a=28..32)
  const int laneM = lane & 15, laneH = lane >> 4;

  // ---------------- init: weight slice -> LDS bf16 (ISA B-operand friendly layout) ----------------
  for (int idx = tid; idx < NA * NN * MCOLS; idx += WGS) {
    int a   = idx >> 12;                               // NN*MCOLS = 4096
    int n   = (idx >> 4) & (NN - 1);
    int col = idx & (MCOLS - 1);
    const float* src = (a < AD) ? &Wa[((size_t)(a * NN + n)) * NN + (m0 + col)]
                                : &Wo[(size_t)n * NN + (m0 + col)];
    if (idx + WGS < NA * NN * MCOLS) __builtin_prefetch(src + (WGS >> 4) * NN, 0, 1);
    W_lds[(a * MCOLS + col) * RSTR + n] = f2bf(*src);
  }
  // ---------------- init: r0 bump (center=128, width=25.6, L2-normalized) ----------------
  if (tid < NN) {
    float d    = fabsf((float)tid - 128.0f);
    float dist = fminf(d, 256.0f - d);
    p_lds[tid] = expf(-dist * dist * (1.0f / (2.0f * 25.6f * 25.6f)));
  }
  __syncthreads();
  if (tid == 0) {
    float ss = 0.0f;
    for (int n = 0; n < NN; ++n) { float v = p_lds[n]; ss += v * v; }
    p_lds[NN] = rsqrtf(ss);
  }
  __syncthreads();
  {
    float inv = p_lds[NN];
    for (int i = tid; i < BLOC * NN; i += WGS) {
      int b = i >> 8, n = i & (NN - 1);
      r0scr[(size_t)(b0 + b) * NN + n] = p_lds[n] * inv;   // identical bits from all writers of this half
    }
  }
  __syncthreads();

  // ---------------- time loop: one grid barrier per step ----------------
  for (int t = 0; t < TT; ++t) {
    // P1a: load r(t-1) f32 -> bf16 LDS tile
    for (int i = tid; i < BLOC * NN; i += WGS) {
      int b = i >> 8, n = i & (NN - 1);
      int bg = b0 + b;
      float f = (t == 0) ? r0scr[(size_t)bg * NN + n]
                         : bump[((size_t)bg * TT + (t - 1)) * NN + n];
      r_lds[b * RSTR + n] = f2bf(f);
    }
    __syncthreads();
    // P1b: deterministic row-sum tree (for J0 term) + coefficients c[b,a] (a==32 -> J1 for Wo)
    {
      int b = tid >> 4, ch = tid & 15;
      float s = 0.0f;
      #pragma unroll
      for (int k = 0; k < 16; ++k) s += bf2f(r_lds[b * RSTR + ch * 16 + k]);
      red_lds[b * 16 + ch] = s;
    }
    for (int i = tid; i < BLOC * NA; i += WGS) {
      int b = i / NA, a = i - b * NA;
      c_lds[i] = (a < AD) ? action[((size_t)(b0 + b) * TT + t) * AD + a] : J1C;
    }
    __syncthreads();
    if (tid < BLOC) {
      float s = 0.0f;
      #pragma unroll
      for (int ch = 0; ch < 16; ++ch) s += red_lds[tid * 16 + ch];
      s_lds[tid] = s;
    }

    // P2: WMMA bf16 GEMM: partial[tile][ag] = sum_{a in group} c[b,a] * (r @ W[a])  (f32 accum)
    {
      // cache the wave's 8 A-fragments (16 batches x K=32 each) in VGPRs; reused for every a
      v4u a0[8], a1[8];
      const unsigned short* rrow = r_lds + (tile * 16 + laneM) * RSTR;
      #pragma unroll
      for (int kf = 0; kf < 8; ++kf) {
        const unsigned short* p = rrow + kf * 32 + laneH * 8;  // lanes16-31 start at K=8 / K=24
        a0[kf] = *(const v4u*)p;            // K +0..7
        a1[kf] = *(const v4u*)(p + 16);     // K +16..23
      }
      v8f acc = {};
      const int aEnd = (ag == 7) ? NA : ag * 4 + 4;
      for (int a = ag * 4; a < aEnd; ++a) {
        v8f part = {};
        const unsigned short* wb = W_lds + (a * MCOLS + laneM) * RSTR + laneH * 8;
        #pragma unroll
        for (int kf = 0; kf < 8; ++kf) {
          ABFrag fa, fb;
          fa.u[0] = a0[kf]; fa.u[1] = a1[kf];
          fb.u[0] = *(const v4u*)(wb + kf * 32);       // B: lanes0-15 K0..7,16..23, lanes16-31 K8..15,24..31
          fb.u[1] = *(const v4u*)(wb + kf * 32 + 16);
          part = __builtin_amdgcn_wmma_f32_16x16x32_bf16(
                   false, fa.v, false, fb.v, (short)0, part, false, false);
        }
        #pragma unroll
        for (int j = 0; j < 8; ++j) {                  // D-frag: lane, VGPR j -> M = j + 8*laneH
          float cv = c_lds[(tile * 16 + j + 8 * laneH) * NA + a];
          acc[j] = fmaf(cv, part[j], acc[j]);
        }
      }
      float* pd = p_lds + (tile * 8 + ag) * 256;
      #pragma unroll
      for (int j = 0; j < 8; ++j) pd[(j + 8 * laneH) * 16 + laneM] = acc[j];
    }
    __syncthreads();

    // P3: reduce the 8 a-group partials, add J0 term, ReLU, leaky update, emit bump(t)
    {
      int b = tid >> 4;                                // local batch 0..31
      int nc = tid & 15;                               // local column
      float rec = J0C * s_lds[b];
      const float* ps = p_lds + ((b >> 4) * 8) * 256 + (b & 15) * 16 + nc;
      #pragma unroll
      for (int a8 = 0; a8 < 8; ++a8) rec += ps[a8 * 256];
      int bg = b0 + b, mg = m0 + nc;
      float ro = (t == 0) ? r0scr[(size_t)bg * NN + mg]
                          : bump[((size_t)bg * TT + (t - 1)) * NN + mg];
      float rn = (1.0f - ALPHAC) * ro + ALPHAC * fmaxf(rec, 0.0f);
      bump[((size_t)bg * TT + t) * NN + mg] = rn;
    }
    grid_sync(counter, (unsigned)(NWG * (t + 1)));
  }

  // ---------------- final phase: r_delta7 = bump @ cos(theta_i - theta_j)  (rank-2, exact) ----------------
  if (tid < NN) {
    float th = (float)tid * (6.283185307179586f / 256.0f);
    p_lds[tid]      = cosf(th);
    p_lds[NN + tid] = sinf(th);
  }
  __syncthreads();
  const float* ct = p_lds;
  const float* st = p_lds + NN;
  for (int row = g * 16 + wave; row < BB * TT; row += NWG * 16) {   // row = b*T + t
    float C = 0.0f, S = 0.0f;
    #pragma unroll
    for (int mi = 0; mi < 8; ++mi) {
      int m = mi * 32 + lane;
      float f = bump[(size_t)row * NN + m];
      C = fmaf(f, ct[m], C);
      S = fmaf(f, st[m], S);
    }
    #pragma unroll
    for (int off = 16; off > 0; off >>= 1) {
      C += __shfl_xor(C, off, 32);
      S += __shfl_xor(S, off, 32);
    }
    float vals[8], mx = -3.4e38f;
    #pragma unroll
    for (int mi = 0; mi < 8; ++mi) {
      int m = mi * 32 + lane;
      vals[mi] = fmaf(C, ct[m], S * st[m]);
      mx = fmaxf(mx, vals[mi]);
    }
    #pragma unroll
    for (int off = 16; off > 0; off >>= 1) mx = fmaxf(mx, __shfl_xor(mx, off, 32));
    float inv = 1.0f / mx;
    #pragma unroll
    for (int mi = 0; mi < 8; ++mi)
      hist[(size_t)row * NN + mi * 32 + lane] = vals[mi] * inv;
  }
}

extern "C" void kernel_launch(void* const* d_in, const int* in_sizes, int n_in,
                              void* d_out, int out_size, void* d_ws, size_t ws_size,
                              hipStream_t stream) {
  (void)in_sizes; (void)n_in; (void)out_size; (void)ws_size;
  const float* action = (const float*)d_in[0];
  const float* Wo     = (const float*)d_in[1];
  const float* Wa     = (const float*)d_in[2];
  float* out          = (float*)d_out;
  unsigned* counter   = (unsigned*)d_ws;

  // barrier counter must start at 0 every call (monotonic within a call)
  hipMemsetAsync(d_ws, 0, sizeof(unsigned), stream);
  // allow > default dynamic LDS (we use ~311 KB of the 320 KB WGP LDS)
  (void)hipFuncSetAttribute((const void*)ring_attractor_kernel,
                            hipFuncAttributeMaxDynamicSharedMemorySize,
                            (int)SMEM_BYTES);
  ring_attractor_kernel<<<NWG, WGS, SMEM_BYTES, stream>>>(action, Wo, Wa, out, counter);
}
